// SwinBlock_79929341378987
// MI455X (gfx1250) — matrix operand order repaired and checked by reading, not compiled, not run on hardware
//
#include <hip/hip_runtime.h>
#include <math.h>

// ---- Swin block constants (match reference) ----
#define DIM    192
#define HEADS  6
#define HD     32
#define NTOK   64          // tokens per window (8*8)
#define SHIFTV 4
#define HIDDEN 768

typedef __attribute__((ext_vector_type(16))) _Float16 v16h;
typedef __attribute__((ext_vector_type(8)))  _Float16 v8h;
typedef __attribute__((ext_vector_type(4)))  _Float16 v4h;
typedef __attribute__((ext_vector_type(8)))  float    v8f;

// ---------------------------------------------------------------------------
// WMMA fragment helpers (CDNA5 wave32 layouts, CDNA5 ISA §7.12.2)
// A (16x32 f16) row-major [m][k]: lane m=L&15, grp=L>>4;
//     halves 0..7 -> k = kb+8*grp+0..7 ; halves 8..15 -> +16  (two b128 loads)
// B (32x16 f16) given as B^T [n][k] row-major: lane n=L&15, grp=L>>4;
//     halves 0..7 -> k = kb+8*grp+0..7 ; halves 8..15 -> +16  (two b128 loads)
// C/D (16x16 f32): lane col n = L&15; VGPR r -> row m = r + 8*(L>>4)
// ---------------------------------------------------------------------------
__device__ __forceinline__ v16h frag_combine(const _Float16* p0, const _Float16* p1) {
  v8h a = *(const v8h*)p0;
  v8h b = *(const v8h*)p1;
  v16h r;
#pragma unroll
  for (int i = 0; i < 8; ++i) { r[i] = a[i]; r[i + 8] = b[i]; }
  return r;
}

__device__ __forceinline__ v16h load_fragA(const _Float16* base, int pitch, int kb) {
  int lane = threadIdx.x & 31;
  const _Float16* row = base + (lane & 15) * pitch + kb + ((lane >> 4) << 3);
  return frag_combine(row, row + 16);
}

__device__ __forceinline__ v16h load_fragB(const _Float16* baseT, int pitch, int kb) {
  int lane = threadIdx.x & 31;
  const _Float16* row = baseT + (lane & 15) * pitch + kb + ((lane >> 4) << 3);
  return frag_combine(row, row + 16);
}

__device__ __forceinline__ v8f wmma16(v16h a, v16h b, v8f c) {
  return __builtin_amdgcn_wmma_f32_16x16x32_f16(false, a, false, b, (short)0, c,
                                                false, false);
}

// ---------------------------------------------------------------------------
// Kernel 0: fp32 -> f16 weight conversion into workspace
// ---------------------------------------------------------------------------
__global__ void k_f32_to_f16(const float* __restrict__ src,
                             _Float16* __restrict__ dst, int n) {
  int i = blockIdx.x * blockDim.x + threadIdx.x;
  if (i < n) dst[i] = (_Float16)src[i];
}

// per-token LayerNorm into LDS (f16, pitch 200), float4-vectorized
__device__ __forceinline__ void ln_token(const float* __restrict__ px,
                                         const float* __restrict__ g,
                                         const float* __restrict__ bta,
                                         _Float16* __restrict__ dst) {
  const float4* p4 = (const float4*)px;
  float s1 = 0.f, s2 = 0.f;
#pragma unroll 4
  for (int i = 0; i < DIM / 4; ++i) {
    float4 v = p4[i];
    s1 += v.x + v.y + v.z + v.w;
    s2 += v.x * v.x + v.y * v.y + v.z * v.z + v.w * v.w;
  }
  float mu  = s1 * (1.f / DIM);
  float inv = rsqrtf(s2 * (1.f / DIM) - mu * mu + 1e-5f);
  const float4* g4 = (const float4*)g;
  const float4* b4 = (const float4*)bta;
#pragma unroll 4
  for (int i = 0; i < DIM / 4; ++i) {
    float4 v = p4[i], gg = g4[i], bb = b4[i];
    v4h o;
    o[0] = (_Float16)((v.x - mu) * inv * gg.x + bb.x);
    o[1] = (_Float16)((v.y - mu) * inv * gg.y + bb.y);
    o[2] = (_Float16)((v.z - mu) * inv * gg.z + bb.z);
    o[3] = (_Float16)((v.w - mu) * inv * gg.w + bb.w);
    *(v4h*)(dst + 4 * i) = o;
  }
}

// ---------------------------------------------------------------------------
// Kernel 1: fused LN1 + shift + windowed attention + proj + residual
// One 4-wave workgroup per 8x8 window. All GEMMs computed transposed
// (C^T = W * X^T): A = weight rows from global, B = LDS activation tile
// cached in registers, epilogue stores contiguous per lane.
// ---------------------------------------------------------------------------
__global__ void __launch_bounds__(128)
k_swin_attn(const float* __restrict__ x,
            const float* __restrict__ ln1_g, const float* __restrict__ ln1_b,
            const _Float16* __restrict__ qkvW, const float* __restrict__ qkv_b,
            const _Float16* __restrict__ projW, const float* __restrict__ proj_b,
            const float* __restrict__ rel_table,
            float* __restrict__ x2) {
  extern __shared__ char smem[];
  _Float16* xln = (_Float16*)smem;          // 64 x (192+8)   LN1(x) [token][ch]
  _Float16* qT  = xln + 64 * 200;           // 64 x (192+8)   Q [token][ch] (A for S)
  _Float16* kT  = qT  + 64 * 200;           // 64 x (192+8)   K [token][ch] (B^T for S)
  _Float16* vT  = kT  + 64 * 200;           // 192 x (64+8)   V^T [ch][token] (A for O^T)
  float*    sS  = (float*)(vT + 192 * 72);  // 4 x 16 x 64    per-wave scores
  _Float16* pP  = (_Float16*)(sS + 4096);   // 4 x 16 x 64    softmax(P) [qtok][ktok]
  _Float16* oO  = pP + 4096;                // 64 x (192+8)   attn out [token][ch]

  const int w    = blockIdx.x;
  const int b    = w >> 6;
  const int wr   = (w >> 3) & 7;
  const int wc   = w & 7;
  const int tid  = threadIdx.x;
  const int wid  = tid >> 5;
  const int lane = tid & 31;
  const int grp  = lane >> 4;

  // ---- LN1 + cyclic shift load (64 tokens, one thread each) ----
  if (tid < NTOK) {
    int r = tid >> 3, cc = tid & 7;
    int hh = (wr * 8 + r + SHIFTV) & 63;
    int ww = (wc * 8 + cc + SHIFTV) & 63;
    ln_token(x + (((size_t)b * 64 + hh) * 64 + ww) * DIM, ln1_g, ln1_b,
             xln + tid * 200);
  }
  __syncthreads();

  // ---- QKV^T GEMM: C^T(576x64) = Wqkv(576x192) @ xln^T. Wave = 16 tokens. ----
  {
    v16h bx[6];
#pragma unroll
    for (int kt = 0; kt < 6; ++kt)
      bx[kt] = load_fragB(xln + wid * 16 * 200, 200, kt * 32);
    const int tok = wid * 16 + (lane & 15);
    for (int mt = 0; mt < 36; ++mt) {
      if (mt + 1 < 36) __builtin_prefetch(qkvW + (size_t)(mt + 1) * 16 * DIM, 0, 1);
      v8f acc = {};
#pragma unroll
      for (int kt = 0; kt < 6; ++kt)
        acc = wmma16(load_fragA(qkvW + (size_t)mt * 16 * DIM, DIM, kt * 32),
                     bx[kt], acc);
      const int chb = mt * 16 + grp * 8;          // 8 contiguous out channels
      if (mt < 12) {                              // Q
        v8h pk;
#pragma unroll
        for (int rr = 0; rr < 8; ++rr) pk[rr] = (_Float16)(acc[rr] + qkv_b[chb + rr]);
        *(v8h*)(qT + tok * 200 + chb) = pk;
      } else if (mt < 24) {                       // K
        v8h pk;
#pragma unroll
        for (int rr = 0; rr < 8; ++rr) pk[rr] = (_Float16)(acc[rr] + qkv_b[chb + rr]);
        *(v8h*)(kT + tok * 200 + (chb - DIM)) = pk;
      } else {                                    // V, stored transposed [ch][tok]
        int ch = chb - 2 * DIM;
#pragma unroll
        for (int rr = 0; rr < 8; ++rr)
          vT[(ch + rr) * 72 + tok] = (_Float16)(acc[rr] + qkv_b[chb + rr]);
      }
    }
  }
  __syncthreads();

  // ---- Attention. Wave owns query rows [16*wid, 16*wid+16). ----
  const float scale = 0.17677669529663687f;   // 1/sqrt(32)
  const int hb = (wr == 7), wb = (wc == 7);
  // (n,m) rel-pos indices and shift-mask adds are head-independent: precompute
  int   relidx[4][8];
  float madd[4][8];
#pragma unroll
  for (int nt = 0; nt < 4; ++nt) {
    int m  = nt * 16 + (lane & 15);
    int mr = m >> 3, mc = m & 7;
    int mcode = (hb ? ((mr < 4) ? 3 : 6) : 0) + (wb ? ((mc < 4) ? 1 : 2) : 0);
#pragma unroll
    for (int rr = 0; rr < 8; ++rr) {
      int n  = wid * 16 + rr + grp * 8;
      int nr = n >> 3, nc = n & 7;
      relidx[nt][rr] = ((nr - mr + 7) * 15 + (nc - mc + 7)) * HEADS;
      int ncode = (hb ? ((nr < 4) ? 3 : 6) : 0) + (wb ? ((nc < 4) ? 1 : 2) : 0);
      madd[nt][rr] = (ncode != mcode) ? -100.f : 0.f;
    }
  }

  for (int h = 0; h < HEADS; ++h) {
    // S stripe (16 x 64) = q_stripe @ k^T
    v16h aQ = load_fragA(qT + wid * 16 * 200 + h * 32, 200, 0);
    for (int nt = 0; nt < 4; ++nt) {
      v8f s = {};
      s = wmma16(aQ, load_fragB(kT + nt * 16 * 200 + h * 32, 200, 0), s);
      int m = nt * 16 + (lane & 15);
#pragma unroll
      for (int rr = 0; rr < 8; ++rr) {
        float v = s[rr] * scale + rel_table[relidx[nt][rr] + h] + madd[nt][rr];
        sS[wid * 1024 + (rr + grp * 8) * 64 + m] = v;
      }
    }
    // softmax: 2 lanes per row (32 cols each), combine via shfl_xor(16)
    {
      int row = lane & 15;
      float* rp = sS + wid * 1024 + row * 64 + grp * 32;
      float mx = -1e30f;
#pragma unroll 8
      for (int m2 = 0; m2 < 32; ++m2) mx = fmaxf(mx, rp[m2]);
      mx = fmaxf(mx, __shfl_xor(mx, 16, 32));
      float sum = 0.f;
#pragma unroll 8
      for (int m2 = 0; m2 < 32; ++m2) { float e = __expf(rp[m2] - mx); rp[m2] = e; sum += e; }
      sum += __shfl_xor(sum, 16, 32);
      float rinv = 1.f / sum;
      _Float16* pr = pP + wid * 1024 + row * 64 + grp * 32;
#pragma unroll 8
      for (int m2 = 0; m2 < 32; ++m2) pr[m2] = (_Float16)(rp[m2] * rinv);
    }
    // O^T (32ch x 16tok) = V_h^T @ P^T : A = vT rows, B^T = pP rows
    for (int mt = 0; mt < 2; ++mt) {
      v8f o = {};
#pragma unroll
      for (int kt = 0; kt < 2; ++kt)
        o = wmma16(load_fragA(vT + (h * 32 + mt * 16) * 72, 72, kt * 32),
                   load_fragB(pP + wid * 1024, 64, kt * 32), o);
      int tok = wid * 16 + (lane & 15);
      v8h pk;
#pragma unroll
      for (int rr = 0; rr < 8; ++rr) pk[rr] = (_Float16)o[rr];
      *(v8h*)(oO + tok * 200 + h * 32 + mt * 16 + grp * 8) = pk;
    }
  }
  __syncthreads();

  // ---- proj^T GEMM + reverse shift + residual -> x2 (contiguous stores) ----
  {
    v16h bo[6];
#pragma unroll
    for (int kt = 0; kt < 6; ++kt)
      bo[kt] = load_fragB(oO + wid * 16 * 200, 200, kt * 32);
    const int tok = wid * 16 + (lane & 15);
    const int r = tok >> 3, cc = tok & 7;
    const int hh = (wr * 8 + r + SHIFTV) & 63;
    const int ww = (wc * 8 + cc + SHIFTV) & 63;
    const size_t rowg = (((size_t)b * 64 + hh) * 64 + ww) * DIM;
    for (int mt = 0; mt < 12; ++mt) {
      v8f acc = {};
#pragma unroll
      for (int kt = 0; kt < 6; ++kt)
        acc = wmma16(load_fragA(projW + (size_t)mt * 16 * DIM, DIM, kt * 32),
                     bo[kt], acc);
      int chb = mt * 16 + grp * 8;
      const float* pxr = x + rowg + chb;
      float* po = x2 + rowg + chb;
#pragma unroll
      for (int rr = 0; rr < 8; ++rr)
        po[rr] = pxr[rr] + acc[rr] + proj_b[chb + rr];
    }
  }
}

// ---------------------------------------------------------------------------
// Kernel 2: fused LN2 + MLP (192->768 GELU ->192) + residual, in-place on x2.
// 64 tokens per 4-wave workgroup; both GEMMs transposed.
// ---------------------------------------------------------------------------
__global__ void __launch_bounds__(128)
k_swin_mlp(float* __restrict__ x2,
           const float* __restrict__ ln2_g, const float* __restrict__ ln2_b,
           const _Float16* __restrict__ w1, const float* __restrict__ b1,
           const _Float16* __restrict__ w2, const float* __restrict__ b2) {
  extern __shared__ char smem[];
  _Float16* zln = (_Float16*)smem;   // 64 x (192+8)  [token][ch]
  _Float16* h1  = zln + 64 * 200;    // 64 x (768+8)  [token][hidden]

  const int tid  = threadIdx.x;
  const int wid  = tid >> 5;
  const int lane = tid & 31;
  const int grp  = lane >> 4;
  const size_t tbase = (size_t)blockIdx.x * 64;

  if (tid < NTOK)
    ln_token(x2 + (tbase + tid) * DIM, ln2_g, ln2_b, zln + tid * 200);
  __syncthreads();

  const int tok = wid * 16 + (lane & 15);

  // GEMM1^T: h1^T(768x64) = W1 @ zln^T, + bias + exact GELU
  {
    v16h bz[6];
#pragma unroll
    for (int kt = 0; kt < 6; ++kt)
      bz[kt] = load_fragB(zln + wid * 16 * 200, 200, kt * 32);
    for (int mt = 0; mt < 48; ++mt) {
      if (mt + 1 < 48) __builtin_prefetch(w1 + (size_t)(mt + 1) * 16 * DIM, 0, 1);
      v8f acc = {};
#pragma unroll
      for (int kt = 0; kt < 6; ++kt)
        acc = wmma16(load_fragA(w1 + (size_t)mt * 16 * DIM, DIM, kt * 32),
                     bz[kt], acc);
      int chb = mt * 16 + grp * 8;
      v8h pk;
#pragma unroll
      for (int rr = 0; rr < 8; ++rr) {
        float v = acc[rr] + b1[chb + rr];
        v = 0.5f * v * (1.f + erff(v * 0.70710678118654752f));
        pk[rr] = (_Float16)v;
      }
      *(v8h*)(h1 + tok * 776 + chb) = pk;      // contiguous ds_store_b128
    }
  }

  // GEMM2^T: out^T(192x64) = W2 @ h1^T. kt-outer, 12 accumulators,
  // each LDS B fragment loaded exactly once. (h1 rows are wave-local.)
  {
    v8f acc[12];
#pragma unroll
    for (int mt = 0; mt < 12; ++mt) acc[mt] = (v8f){};
    for (int kt = 0; kt < 24; ++kt) {
      v16h bh = load_fragB(h1 + wid * 16 * 776, 776, kt * 32);
#pragma unroll
      for (int mt = 0; mt < 12; ++mt)
        acc[mt] = wmma16(load_fragA(w2 + (size_t)mt * 16 * HIDDEN, HIDDEN, kt * 32),
                         bh, acc[mt]);
    }
    float* prow = x2 + (tbase + tok) * DIM;
#pragma unroll
    for (int mt = 0; mt < 12; ++mt) {
      int chb = mt * 16 + grp * 8;
#pragma unroll
      for (int rr = 0; rr < 8; ++rr)
        prow[chb + rr] = prow[chb + rr] + acc[mt][rr] + b2[chb + rr];
    }
  }
}

// ---------------------------------------------------------------------------
extern "C" void kernel_launch(void* const* d_in, const int* in_sizes, int n_in,
                              void* d_out, int out_size, void* d_ws, size_t ws_size,
                              hipStream_t stream) {
  (void)in_sizes; (void)n_in; (void)out_size; (void)ws_size;
  const float* x      = (const float*)d_in[0];
  const float* ln1_g  = (const float*)d_in[1];
  const float* ln1_b  = (const float*)d_in[2];
  const float* qkv_w  = (const float*)d_in[3];
  const float* qkv_b  = (const float*)d_in[4];
  const float* proj_w = (const float*)d_in[5];
  const float* proj_b = (const float*)d_in[6];
  const float* rel    = (const float*)d_in[7];
  const float* ln2_g  = (const float*)d_in[8];
  const float* ln2_b  = (const float*)d_in[9];
  const float* mlp_w1 = (const float*)d_in[10];
  const float* mlp_b1 = (const float*)d_in[11];
  const float* mlp_w2 = (const float*)d_in[12];
  const float* mlp_b2 = (const float*)d_in[13];

  // f16 weight staging in workspace (~0.9 MB)
  _Float16* qkvW  = (_Float16*)d_ws;
  _Float16* projW = qkvW  + 576 * 192;
  _Float16* w1h   = projW + 192 * 192;
  _Float16* w2h   = w1h   + 768 * 192;

  int n;
  n = 576 * 192; k_f32_to_f16<<<(n + 255) / 256, 256, 0, stream>>>(qkv_w, qkvW, n);
  n = 192 * 192; k_f32_to_f16<<<(n + 255) / 256, 256, 0, stream>>>(proj_w, projW, n);
  n = 768 * 192; k_f32_to_f16<<<(n + 255) / 256, 256, 0, stream>>>(mlp_w1, w1h, n);
  n = 192 * 768; k_f32_to_f16<<<(n + 255) / 256, 256, 0, stream>>>(mlp_w2, w2h, n);

  float* x2 = (float*)d_out;

  size_t shA = (size_t)(64 * 200 * 3 + 192 * 72 + 4096 + 64 * 200) * 2 + 4096 * 4; // 154624 B
  size_t shM = (size_t)(64 * 200 + 64 * 776) * 2;                                  // 124928 B
  hipFuncSetAttribute(reinterpret_cast<const void*>(&k_swin_attn),
                      hipFuncAttributeMaxDynamicSharedMemorySize, (int)shA);
  hipFuncSetAttribute(reinterpret_cast<const void*>(&k_swin_mlp),
                      hipFuncAttributeMaxDynamicSharedMemorySize, (int)shM);

  k_swin_attn<<<2048, 128, shA, stream>>>(x, ln1_g, ln1_b, qkvW, qkv_b,
                                          projW, proj_b, rel, x2);
  k_swin_mlp<<<2048, 128, shM, stream>>>(x2, ln2_g, ln2_b, w1h, mlp_b1, w2h, mlp_b2);
}
